// GQAttention_72988674228289
// MI455X (gfx1250) — compile-verified, matrix operand order checked by reading, not run on hardware
//
#include <hip/hip_runtime.h>

#define BB 2
#define SS 2048
#define EE 2048
#define HH 16
#define GG 4
#define HDD 128
#define GDD 512

typedef __attribute__((ext_vector_type(16))) __bf16 v16bf;
typedef __attribute__((ext_vector_type(8)))  __bf16 v8bf;
typedef __attribute__((ext_vector_type(8)))  float  v8f;

__device__ __forceinline__ v8f wmma_bf16(v16bf a, v16bf b, v8f c) {
  // (neg_a, A, neg_b, B, c_mod, C, reuse_a, reuse_b)
  return __builtin_amdgcn_wmma_f32_16x16x32_bf16(false, a, false, b, (short)0, c,
                                                 false, false);
}

// DPP lane-permute of a float (row_mask=0xF, bank_mask=0xF, bound_ctrl=1)
#define DPPF(x, ctrl) \
  __int_as_float(__builtin_amdgcn_update_dpp(0, __float_as_int(x), (ctrl), 0xF, 0xF, true))

// Max across each 16-lane half (wave32 has two DPP rows of 16):
// ror:8, ror:4 -> each lane holds max of its mod-4 residue class (4 members),
// quad_perm xor2, xor1 -> merge the 4 classes.
__device__ __forceinline__ float rowmax16(float x) {
  x = fmaxf(x, DPPF(x, 0x128));  // row_ror:8
  x = fmaxf(x, DPPF(x, 0x124));  // row_ror:4
  x = fmaxf(x, DPPF(x, 0x4E));   // quad_perm [2,3,0,1]  (xor 2)
  x = fmaxf(x, DPPF(x, 0xB1));   // quad_perm [1,0,3,2]  (xor 1)
  return x;
}

// A/B fragment for v_wmma_f32_16x16x32_bf16:
// lane L: row = L&15, half = L>>4; elems 0..7 -> K = half*8 + e,
// elems 8..15 -> K = 16 + half*8 + (e-8). Two contiguous 16B loads.
__device__ __forceinline__ v16bf frag_from_bf16(const __bf16* base, long stride,
                                                int row, int k0, int half) {
  const __bf16* p = base + (long)row * stride + k0 + half * 8;
  v8bf lo = *(const v8bf*)(p);
  v8bf hi = *(const v8bf*)(p + 16);
  v16bf r;
#pragma unroll
  for (int i = 0; i < 8; ++i) { r[i] = lo[i]; r[i + 8] = hi[i]; }
  return r;
}

__device__ __forceinline__ v16bf frag_from_f32(const float* base, long stride,
                                               int row, int k0, int half) {
  const float* p = base + (long)row * stride + k0 + half * 8;
  v8f lo = *(const v8f*)(p);
  v8f hi = *(const v8f*)(p + 16);
  v8bf lb = __builtin_convertvector(lo, v8bf);
  v8bf hb = __builtin_convertvector(hi, v8bf);
  v16bf r;
#pragma unroll
  for (int i = 0; i < 8; ++i) { r[i] = lb[i]; r[i + 8] = hb[i]; }
  return r;
}

// wt[n*K + k] = bf16(w[k*N + n])   (reads coalesced)
__global__ void transpose_bf16_kernel(const float* __restrict__ w,
                                      __bf16* __restrict__ wt, int K, int N) {
  long i = (long)blockIdx.x * blockDim.x + threadIdx.x;
  long total = (long)K * N;
  if (i >= total) return;
  int n = (int)(i % N);
  int k = (int)(i / N);
  wt[(long)n * K + k] = (__bf16)w[i];
}

// C = A(MxK) @ Wt^T + bias.  Wt is [N][K] bf16.
// Each wave computes a 16x64 output tile (one A fragment feeds 4 B tiles),
// software-pipelined (double-buffered fragments) over K.
// MODE 0: bf16 out, head-major [bidx][n/HD][srow][n%HD]
// MODE 1: bf16 out, transposed  [bidx][n/HD][n%HD][srow]   (for V^T)
// MODE 2: f32 out, row-major [m][n]
// ABF16:  A is bf16 (else f32, converted on load)
template <int MODE, bool ABF16>
__global__ void __launch_bounds__(256, 1)
gemm_wmma_kernel(const void* __restrict__ Asrc,
                 const __bf16* __restrict__ Wt,
                 const float* __restrict__ bias,
                 int M, int N, int K,
                 __bf16* __restrict__ outb,
                 float* __restrict__ outf,
                 int HN, int Sdim, int HDdim) {
  const float*  Af = (const float*)Asrc;
  const __bf16* Ab = (const __bf16*)Asrc;
  const int lane = threadIdx.x & 31;
  const int wave = threadIdx.x >> 5;
  const int half = lane >> 4;
  const int nl = lane & 15;
  const int m0 = blockIdx.y * 16;
  const int n0 = blockIdx.x * 512 + wave * 64;
  const int arow = m0 + nl;

  v8f c[4] = {};

  // prologue: fragments for k0 = 0
  v16bf a_c = ABF16 ? frag_from_bf16(Ab, K, arow, 0, half)
                    : frag_from_f32(Af, K, arow, 0, half);
  v16bf b_c[4];
#pragma unroll
  for (int t = 0; t < 4; ++t)
    b_c[t] = frag_from_bf16(Wt, K, n0 + t * 16 + nl, 0, half);

  for (int k0 = 0; k0 < K - 32; k0 += 32) {
    const int kn = k0 + 32;
    // prefetch a few K-steps ahead (global_prefetch_b8)
    if (kn + 256 < K) {
      __builtin_prefetch((const void*)(Ab + (long)arow * K + kn + 256), 0, 1);
      __builtin_prefetch((const void*)(Wt + (long)(n0 + nl) * K + kn + 256), 0, 1);
    }
    // issue next tile's loads before this tile's WMMAs
    v16bf a_n = ABF16 ? frag_from_bf16(Ab, K, arow, kn, half)
                      : frag_from_f32(Af, K, arow, kn, half);
    v16bf b_n[4];
#pragma unroll
    for (int t = 0; t < 4; ++t)
      b_n[t] = frag_from_bf16(Wt, K, n0 + t * 16 + nl, kn, half);
#pragma unroll
    for (int t = 0; t < 4; ++t) c[t] = wmma_bf16(a_c, b_c[t], c[t]);
    a_c = a_n;
#pragma unroll
    for (int t = 0; t < 4; ++t) b_c[t] = b_n[t];
  }
#pragma unroll
  for (int t = 0; t < 4; ++t) c[t] = wmma_bf16(a_c, b_c[t], c[t]);

#pragma unroll
  for (int t = 0; t < 4; ++t) {
    const float bi = bias[n0 + t * 16 + nl];
    const long n = n0 + t * 16 + nl;
#pragma unroll
    for (int r = 0; r < 8; ++r) {
      const long m = m0 + r + 8 * half;   // C layout: row = r + 8*half
      const float v = c[t][r] + bi;
      if (MODE == 2) {
        outf[m * (long)N + n] = v;
      } else {
        const long bidx = m / Sdim, srow = m % Sdim;
        const long hh = n / HDdim, d = n % HDdim;
        const long addr = (MODE == 0)
            ? ((bidx * HN + hh) * Sdim + srow) * HDdim + d
            : ((bidx * HN + hh) * HDdim + d) * Sdim + srow;
        outb[addr] = (__bf16)v;
      }
    }
  }
}

// Flash attention: one wave per 16-query tile, online softmax over 32-key chunks.
// Row max via DPP (VALU-only); row sum via ones-matrix WMMA accumulator.
__global__ void __launch_bounds__(256, 1)
attn_wmma_kernel(const __bf16* __restrict__ Qb,
                 const __bf16* __restrict__ Kb,
                 const __bf16* __restrict__ Vt,
                 const int* __restrict__ mask,
                 __bf16* __restrict__ attn) {
  __shared__ __bf16 plds[8][16 * 32];
  const int lane = threadIdx.x & 31;
  const int wave = threadIdx.x >> 5;
  const int half = lane >> 4;
  const int nl = lane & 15;

  const int t = blockIdx.x * 8 + wave;
  const int qt = t & (SS / 16 - 1);
  const int bh = t / (SS / 16);
  const int b = bh / HH;
  const int h = bh % HH;
  const int g = h % GG;          // tile(k,(1,NGH,1,1)): head h -> group h%G
  const int q0 = qt * 16;

  const __bf16* Qp = Qb + ((long)(b * HH + h) * SS) * HDD;
  const __bf16* Kp = Kb + ((long)(b * GG + g) * SS) * HDD;
  const __bf16* Vp = Vt + ((long)(b * GG + g) * HDD) * SS;   // [HD][S]
  const int* mp = mask + ((long)b * SS + q0) * SS;
  const float scale = 0.08838834764831845f;                  // 1/sqrt(128)

  v16bf qa[4];
#pragma unroll
  for (int kk = 0; kk < 4; ++kk)
    qa[kk] = frag_from_bf16(Qp, HDD, q0 + nl, kk * 32, half);

  v16bf ones;
#pragma unroll
  for (int i = 0; i < 16; ++i) ones[i] = (__bf16)1.0f;

  float mrun[8];
  v8f acc[8] = {};
  v8f accl = {};                 // running softmax denominator per row
#pragma unroll
  for (int r = 0; r < 8; ++r) mrun[r] = -__builtin_inff();

  for (int j = 0; j < SS; j += 32) {
    v8f c0 = {};
    v8f c1 = {};
#pragma unroll
    for (int kk = 0; kk < 4; ++kk) {
      v16bf b0 = frag_from_bf16(Kp, HDD, j + nl, kk * 32, half);
      v16bf b1 = frag_from_bf16(Kp, HDD, j + 16 + nl, kk * 32, half);
      c0 = wmma_bf16(qa[kk], b0, c0);
      c1 = wmma_bf16(qa[kk], b1, c1);
    }

    float al[8], p0[8], p1[8];
#pragma unroll
    for (int r = 0; r < 8; ++r) {
      const int row = r + 8 * half;
      float s0 = c0[r] * scale;
      float s1 = c1[r] * scale;
      if (mp[(long)row * SS + j + nl] == 0) s0 = -10000.f;
      if (mp[(long)row * SS + j + 16 + nl] == 0) s1 = -10000.f;
      const float mx = rowmax16(fmaxf(s0, s1));
      const float nm = fmaxf(mrun[r], mx);
      al[r] = __expf(mrun[r] - nm);
      mrun[r] = nm;
      p0[r] = __expf(s0 - nm);
      p1[r] = __expf(s1 - nm);
    }

#pragma unroll
    for (int r = 0; r < 8; ++r) {
      accl[r] *= al[r];
#pragma unroll
      for (int nc = 0; nc < 8; ++nc) acc[nc][r] *= al[r];
    }

    // Restage P (C layout -> A layout) through per-wave LDS tile.
    __bf16* pl = plds[wave];
#pragma unroll
    for (int r = 0; r < 8; ++r) {
      pl[(r + 8 * half) * 32 + nl] = (__bf16)p0[r];
      pl[(r + 8 * half) * 32 + 16 + nl] = (__bf16)p1[r];
    }
    asm volatile("" ::: "memory");   // same-wave DS ops are in-order
    v16bf pa = frag_from_bf16(pl, 32, nl, 0, half);
    asm volatile("" ::: "memory");

    accl = wmma_bf16(pa, ones, accl);   // row sums of P -> denominator
#pragma unroll
    for (int nc = 0; nc < 8; ++nc) {
      v16bf bv = frag_from_bf16(Vp, SS, nc * 16 + nl, j, half);
      acc[nc] = wmma_bf16(pa, bv, acc[nc]);
    }
  }

  // attn[b][q][h*HD + d] (bf16) feeds the O-projection GEMM
  __bf16* op = attn + ((long)b * SS + q0) * EE + h * HDD;
#pragma unroll
  for (int r = 0; r < 8; ++r) {
    const float inv = 1.0f / accl[r];
#pragma unroll
    for (int nc = 0; nc < 8; ++nc)
      op[(long)(r + 8 * half) * EE + nc * 16 + nl] = (__bf16)(acc[nc][r] * inv);
  }
}

extern "C" void kernel_launch(void* const* d_in, const int* in_sizes, int n_in,
                              void* d_out, int out_size, void* d_ws, size_t ws_size,
                              hipStream_t stream) {
  const float* x  = (const float*)d_in[0];
  const int* mask = (const int*)d_in[1];
  const float* Wq = (const float*)d_in[2];
  const float* bq = (const float*)d_in[3];
  const float* Wk = (const float*)d_in[4];
  const float* bk = (const float*)d_in[5];
  const float* Wv = (const float*)d_in[6];
  const float* bv = (const float*)d_in[7];
  const float* Wo = (const float*)d_in[8];
  const float* bo = (const float*)d_in[9];
  float* out = (float*)d_out;

  char* ws = (char*)d_ws;
  size_t off = 0;
  auto alloc = [&](size_t bytes) {
    void* p = ws + off;
    off = (off + bytes + 255) & ~(size_t)255;
    return p;
  };
  __bf16* WtQ = (__bf16*)alloc((size_t)EE * EE * 2);
  __bf16* WtK = (__bf16*)alloc((size_t)GDD * EE * 2);
  __bf16* WtV = (__bf16*)alloc((size_t)GDD * EE * 2);
  __bf16* WtO = (__bf16*)alloc((size_t)EE * EE * 2);
  __bf16* Qb  = (__bf16*)alloc((size_t)BB * SS * EE * 2);
  __bf16* Kb  = (__bf16*)alloc((size_t)BB * SS * GDD * 2);
  __bf16* Vtb = (__bf16*)alloc((size_t)BB * SS * GDD * 2);
  __bf16* Attn = (__bf16*)alloc((size_t)BB * SS * EE * 2);

  const int TPB = 256;
  const long totEE = (long)EE * EE;
  const long totEG = (long)EE * GDD;
  transpose_bf16_kernel<<<(totEE + TPB - 1) / TPB, TPB, 0, stream>>>(Wq, WtQ, EE, EE);
  transpose_bf16_kernel<<<(totEG + TPB - 1) / TPB, TPB, 0, stream>>>(Wk, WtK, EE, GDD);
  transpose_bf16_kernel<<<(totEG + TPB - 1) / TPB, TPB, 0, stream>>>(Wv, WtV, EE, GDD);
  transpose_bf16_kernel<<<(totEE + TPB - 1) / TPB, TPB, 0, stream>>>(Wo, WtO, EE, EE);

  const int M = BB * SS;
  dim3 gq(EE / 512, M / 16);   // (4, 256)
  dim3 gk(GDD / 512, M / 16);  // (1, 256)
  gemm_wmma_kernel<0, false><<<gq, TPB, 0, stream>>>(x, WtQ, bq, M, EE, EE,
                                                     Qb, nullptr, HH, SS, HDD);
  gemm_wmma_kernel<0, false><<<gk, TPB, 0, stream>>>(x, WtK, bk, M, GDD, EE,
                                                     Kb, nullptr, GG, SS, HDD);
  gemm_wmma_kernel<1, false><<<gk, TPB, 0, stream>>>(x, WtV, bv, M, GDD, EE,
                                                     Vtb, nullptr, GG, SS, HDD);

  const int tiles = BB * HH * (SS / 16);      // 4096 tiles, 8 waves/block
  attn_wmma_kernel<<<tiles / 8, TPB, 0, stream>>>(Qb, Kb, Vtb, mask, Attn);

  gemm_wmma_kernel<2, true><<<gq, TPB, 0, stream>>>(Attn, WtO, bo, M, EE, EE,
                                                    nullptr, out, HH, SS, HDD);
}
